// ImplicitMemoryModule_54408645705792
// MI455X (gfx1250) — compile-verified, matrix operand order checked by reading, not run on hardware
//
#include <hip/hip_runtime.h>
#include <hip/hip_bf16.h>

// ---------- types / helpers -------------------------------------------------
typedef __attribute__((ext_vector_type(16))) __bf16   v16bf;
typedef __attribute__((ext_vector_type(2)))  __bf16   v2bf;
typedef __attribute__((ext_vector_type(8)))  float    v8f;
typedef __attribute__((ext_vector_type(8)))  unsigned v8u;

union ABFrag { v8u u; v16bf bf; };

// fp32x2 -> packed bf16x2 via hardware convert (v_cvt_pk_bf16_f32)
__device__ __forceinline__ unsigned pk_bf16(float a, float b) {
#if __has_builtin(__builtin_amdgcn_cvt_pk_bf16_f32)
  return __builtin_bit_cast(unsigned, __builtin_amdgcn_cvt_pk_bf16_f32(a, b));
#else
  v2bf v;
  v[0] = (__bf16)a;
  v[1] = (__bf16)b;
  return __builtin_bit_cast(unsigned, v);
#endif
}
__device__ __forceinline__ unsigned short bf16b(float x) {
  return __builtin_bit_cast(unsigned short, (__bf16)x);
}
__device__ __forceinline__ v8f wmma_bf16(const ABFrag& a, const ABFrag& b, v8f c) {
  return __builtin_amdgcn_wmma_f32_16x16x32_bf16(false, a.bf, false, b.bf,
                                                 (short)0, c, false, false);
}

// A/B fragment from bf16 row (LDS or global): two 16-byte runs -> 2x b128
__device__ __forceinline__ void loadAB_bf16(ABFrag& d, const unsigned short* row, int kk, int half) {
  const int k0 = kk + half * 8;
  uint4 x = *(const uint4*)(row + k0);
  uint4 y = *(const uint4*)(row + k0 + 16);
  d.u[0] = x.x; d.u[1] = x.y; d.u[2] = x.z; d.u[3] = x.w;
  d.u[4] = y.x; d.u[5] = y.y; d.u[6] = y.z; d.u[7] = y.w;
}

// async global -> LDS 16-byte mover (GV mode, ASYNCcnt-tracked)
__device__ __forceinline__ void async_g2l_b128(unsigned lds_off, const void* gptr) {
  asm volatile("global_load_async_to_lds_b128 %0, %1, off"
               :: "v"(lds_off), "v"(gptr) : "memory");
}
__device__ __forceinline__ void wait_asynccnt0() {
#if __has_builtin(__builtin_amdgcn_s_wait_asynccnt)
  __builtin_amdgcn_s_wait_asynccnt(0);
#else
  asm volatile("s_wait_asynccnt 0" ::: "memory");
#endif
}

#define H_DIM 2048
#define K_DIM 256
#define V_DIM 256
#define N_SLOT 64
#define T_DIM 4096
// LDS row strides (in bf16 elems): multiple of 8 (16B aligned) and
// stride/2 mod 64 banks != 0 -> conflict-free 16-row b128 reads
#define HSTR 2056   // 1028 dwords  (== 4 mod 64)
#define KSTR 264    // 132 dwords   (== 4 mod 64)
#define NSTR 72     // 36 dwords    (== 36 mod 64, odd multiple of 4)

// ---------- kernel 0: one-shot fp32 -> bf16 weight conversion ---------------
__global__ void k0_cvt(const float* __restrict__ src, unsigned short* __restrict__ dst, int n4) {
  const int stride = gridDim.x * blockDim.x;
  for (int i = blockIdx.x * blockDim.x + threadIdx.x; i < n4; i += stride) {
    float4 f = *(const float4*)(src + (size_t)i * 4);
    uint2 p; p.x = pk_bf16(f.x, f.y); p.y = pk_bf16(f.z, f.w);
    *(uint2*)(dst + (size_t)i * 4) = p;
  }
}

// ---------- kernel 1: Q = hid @ Wq^T  (bf16 in, bf16 out) -------------------
// grid.x = (B*T)/16, block = 256 (8 waves, each wave: 2 col tiles of 16)
__global__ void k1_qproj(const float* __restrict__ hid,
                         const unsigned short* __restrict__ Wqb,   // [K,H] bf16
                         unsigned short* __restrict__ Qout) {
  __shared__ __align__(16) unsigned short hidL[16][HSTR];   // 16 x 2048 bf16

  const int tid = threadIdx.x;
  const int lane = tid & 31, wave = tid >> 5;
  const int lm = lane & 15, half = lane >> 4;
  const int rowbase = blockIdx.x * 16;
  const int c0 = wave * 32;                       // cols c0..c0+31

  // stage + convert the 16-row A block once per workgroup (float4 -> 2x bf16x2)
  for (int i = tid; i < 16 * (H_DIM / 4); i += 256) {
    const int n = i >> 9, kp = (i & 511) << 2;
    float4 f = *(const float4*)(hid + (size_t)(rowbase + n) * H_DIM + kp);
    uint2 p; p.x = pk_bf16(f.x, f.y); p.y = pk_bf16(f.z, f.w);
    *(uint2*)&hidL[n][kp] = p;
  }
  __syncthreads();

  v8f acc0 = {}, acc1 = {};
  const unsigned short* brow0 = Wqb + (size_t)(c0 + lm) * H_DIM;
  const unsigned short* brow1 = Wqb + (size_t)(c0 + 16 + lm) * H_DIM;

  #pragma unroll 1
  for (int kk = 0; kk < H_DIM; kk += 32) {
    __builtin_prefetch(brow0 + kk + 512, 0, 1);   // stream B ahead (global_prefetch)
    ABFrag a, b0, b1;
    loadAB_bf16(a, &hidL[lm][0], kk, half);
    loadAB_bf16(b0, brow0, kk, half);
    loadAB_bf16(b1, brow1, kk, half);
    acc0 = wmma_bf16(a, b0, acc0);
    acc1 = wmma_bf16(a, b1, acc1);
  }
  #pragma unroll
  for (int r = 0; r < 8; ++r) {
    const size_t m = (size_t)(rowbase + 8 * half + r);
    Qout[m * K_DIM + c0 + lm]      = bf16b(acc0[r]);
    Qout[m * K_DIM + c0 + 16 + lm] = bf16b(acc1[r]);
  }
}

// ---------- kernel 2: scores -> softmax -> retrieved ------------------------
// grid.x = (B*T)/128, block = 256 (8 waves, each wave owns 16 rows)
__global__ void k2_attn(const unsigned short* __restrict__ Q,
                        const float* __restrict__ keys,
                        const float* __restrict__ vals,
                        const unsigned char* __restrict__ validm,
                        const unsigned char* __restrict__ histm,
                        unsigned short* __restrict__ Rout) {
  __shared__ __align__(16) unsigned short keysL[N_SLOT][KSTR];   // [n][k] bf16
  __shared__ __align__(16) unsigned short valsT[V_DIM][NSTR];    // [v][n] bf16
  __shared__ __align__(16) unsigned short wscr[8][16 * NSTR];    // per-wave weights

  const int tid = threadIdx.x;
  const int lane = tid & 31, wave = tid >> 5;
  const int lm = lane & 15, half = lane >> 4;
  const int blockrow = blockIdx.x * 128;
  const int b = blockrow / T_DIM;
  const int rowbase = blockrow + wave * 16;

  // stage keys (bf16, row-major in k)
  const float* kb = keys + (size_t)b * N_SLOT * K_DIM;
  for (int i = tid; i < N_SLOT * (K_DIM / 4); i += 256) {
    const int n = i >> 6, kp = (i & 63) << 2;
    float4 f = *(const float4*)(kb + n * K_DIM + kp);
    uint2 p; p.x = pk_bf16(f.x, f.y); p.y = pk_bf16(f.z, f.w);
    *(uint2*)&keysL[n][kp] = p;
  }
  // stage values transposed (bf16, [v][n])
  const float* vb = vals + (size_t)b * N_SLOT * V_DIM;
  for (int i = tid; i < V_DIM * (N_SLOT / 2); i += 256) {
    const int v = i >> 5, n = (i & 31) << 1;
    *(unsigned*)&valsT[v][n] = pk_bf16(vb[n * V_DIM + v], vb[(n + 1) * V_DIM + v]);
  }
  __syncthreads();

  // ---- scores: [16 x 64] = Q[16 x 256] x keys^T ----
  v8f sc[4] = {};
  const unsigned short* qrow = Q + (size_t)(rowbase + lm) * K_DIM;
  #pragma unroll 1
  for (int kk = 0; kk < K_DIM; kk += 32) {
    ABFrag a;
    loadAB_bf16(a, qrow, kk, half);
    #pragma unroll
    for (int nt = 0; nt < 4; ++nt) {
      ABFrag bf;
      loadAB_bf16(bf, &keysL[nt * 16 + lm][0], kk, half);
      sc[nt] = wmma_bf16(a, bf, sc[nt]);
    }
  }

  // ---- masked softmax over n (rows live at fixed r across 16-lane halves) --
  int vm[4];
  #pragma unroll
  for (int nt = 0; nt < 4; ++nt) vm[nt] = validm[b * N_SLOT + nt * 16 + lm];
  int anyi = vm[0] | vm[1] | vm[2] | vm[3];
  #pragma unroll
  for (int msk = 1; msk <= 8; msk <<= 1) anyi |= __shfl_xor(anyi, msk, 32);
  const float anyv = anyi ? 1.f : 0.f;

  #pragma unroll
  for (int r = 0; r < 8; ++r) {
    float mx = -3.0e38f;
    #pragma unroll
    for (int nt = 0; nt < 4; ++nt) {
      float v = vm[nt] ? sc[nt][r] : -1.0e9f;
      sc[nt][r] = v;
      mx = fmaxf(mx, v);
    }
    #pragma unroll
    for (int msk = 1; msk <= 8; msk <<= 1) mx = fmaxf(mx, __shfl_xor(mx, msk, 32));
    float sum = 0.f;
    #pragma unroll
    for (int nt = 0; nt < 4; ++nt) {
      float e = __expf(sc[nt][r] - mx);
      sc[nt][r] = e;
      sum += e;
    }
    #pragma unroll
    for (int msk = 1; msk <= 8; msk <<= 1) sum += __shfl_xor(sum, msk, 32);
    const int lr = 8 * half + r;
    const float allowed = histm[rowbase + lr] ? 0.f : 1.f;
    const float scale = anyv * allowed / sum;
    #pragma unroll
    for (int nt = 0; nt < 4; ++nt)
      wscr[wave][lr * NSTR + nt * 16 + lm] = bf16b(sc[nt][r] * scale);
  }

  // ---- retrieved: [16 x 256] = weights[16 x 64] x values ----
  v8f racc[16];
  #pragma unroll
  for (int i = 0; i < 16; ++i) racc[i] = (v8f){};
  const unsigned short* wrow = &wscr[wave][lm * NSTR];
  #pragma unroll
  for (int s = 0; s < 2; ++s) {
    ABFrag a;
    loadAB_bf16(a, wrow, s * 32, half);
    #pragma unroll
    for (int vt = 0; vt < 16; ++vt) {
      ABFrag bf;
      loadAB_bf16(bf, &valsT[vt * 16 + lm][0], s * 32, half);
      racc[vt] = wmma_bf16(a, bf, racc[vt]);
    }
  }
  #pragma unroll
  for (int vt = 0; vt < 16; ++vt) {
    #pragma unroll
    for (int r = 0; r < 8; ++r) {
      const size_t m = (size_t)(rowbase + 8 * half + r);
      Rout[m * V_DIM + vt * 16 + lm] = bf16b(racc[vt][r]);
    }
  }
}

// ---------- kernel 3: merged = R @ Wo^T ; residual ; layernorm --------------
// grid.x = (B*T)/16, block = 256 (8 waves x 16 col tiles = full H=2048)
__global__ void k3_out(const unsigned short* __restrict__ R,
                       const unsigned short* __restrict__ Wob,   // [H,V] bf16
                       const float* __restrict__ hid,
                       const float* __restrict__ gamma,
                       const float* __restrict__ beta,
                       float* __restrict__ out) {
  __shared__ __align__(16) unsigned short RL[16][KSTR];   // 16 x 256 bf16 A block
  __shared__ float redS[8][16], redQ[8][16], muS[16], rsS[16];

  const int tid = threadIdx.x, lane = tid & 31, wave = tid >> 5;
  const int lm = lane & 15, half = lane >> 4;
  const int rowbase = blockIdx.x * 16;
  const int hbase = wave * 256;

  // stage the bf16 A block via the async data mover: 512 x b128 (2 per thread)
  #pragma unroll
  for (int it = 0; it < 2; ++it) {
    const int idx = tid + it * 256;          // b128 index 0..511
    const int n = idx >> 5;                  // 32 x 16B per row
    const int kp = (idx & 31) << 3;          // bf16 elem offset
    async_g2l_b128((unsigned)(size_t)&RL[n][kp],
                   R + (size_t)(rowbase + n) * V_DIM + kp);
  }
  wait_asynccnt0();
  __syncthreads();

  v8f acc[16];
  #pragma unroll
  for (int i = 0; i < 16; ++i) acc[i] = (v8f){};

  #pragma unroll 1
  for (int kk = 0; kk < V_DIM; kk += 32) {
    ABFrag a;
    loadAB_bf16(a, &RL[lm][0], kk, half);
    #pragma unroll
    for (int ht = 0; ht < 16; ++ht) {
      ABFrag bf;
      loadAB_bf16(bf, Wob + (size_t)(hbase + ht * 16 + lm) * V_DIM, kk, half);
      acc[ht] = wmma_bf16(a, bf, acc[ht]);
    }
  }

  // residual add
  #pragma unroll
  for (int ht = 0; ht < 16; ++ht) {
    const int h = hbase + ht * 16 + lm;
    #pragma unroll
    for (int r = 0; r < 8; ++r) {
      const size_t m = (size_t)(rowbase + 8 * half + r);
      acc[ht][r] += hid[m * H_DIM + h];
    }
  }

  // per-row mean / var: reduce over 16 tiles, then 16 lanes, then 8 waves
  float s[8], s2[8];
  #pragma unroll
  for (int r = 0; r < 8; ++r) {
    float a = 0.f, q = 0.f;
    #pragma unroll
    for (int ht = 0; ht < 16; ++ht) {
      float v = acc[ht][r];
      a += v; q += v * v;
    }
    #pragma unroll
    for (int msk = 1; msk <= 8; msk <<= 1) {
      a += __shfl_xor(a, msk, 32);
      q += __shfl_xor(q, msk, 32);
    }
    s[r] = a; s2[r] = q;
  }
  if (lm == 0) {
    #pragma unroll
    for (int r = 0; r < 8; ++r) {
      redS[wave][8 * half + r] = s[r];
      redQ[wave][8 * half + r] = s2[r];
    }
  }
  __syncthreads();
  if (tid < 16) {
    float a = 0.f, q = 0.f;
    #pragma unroll
    for (int w = 0; w < 8; ++w) { a += redS[w][tid]; q += redQ[w][tid]; }
    const float mu = a * (1.f / (float)H_DIM);
    const float var = q * (1.f / (float)H_DIM) - mu * mu;
    muS[tid] = mu;
    rsS[tid] = rsqrtf(var + 1e-5f);
  }
  __syncthreads();

  #pragma unroll
  for (int ht = 0; ht < 16; ++ht) {
    const int h = hbase + ht * 16 + lm;
    const float g = gamma[h], be = beta[h];
    #pragma unroll
    for (int r = 0; r < 8; ++r) {
      const int lr = 8 * half + r;
      const float v = (acc[ht][r] - muS[lr]) * rsS[lr] * g + be;
      out[(size_t)(rowbase + lr) * H_DIM + h] = v;
    }
  }
}

// ---------- launcher --------------------------------------------------------
extern "C" void kernel_launch(void* const* d_in, const int* in_sizes, int n_in,
                              void* d_out, int out_size, void* d_ws, size_t ws_size,
                              hipStream_t stream) {
  const float* hid   = (const float*)d_in[0];           // [B,T,H]
  const float* keys  = (const float*)d_in[1];           // [B,N,K]
  const float* vals  = (const float*)d_in[2];           // [B,N,V]
  const float* Wq    = (const float*)d_in[3];           // [K,H]
  const float* Wo    = (const float*)d_in[4];           // [H,V]
  const float* gamma = (const float*)d_in[5];           // [H]
  const float* beta  = (const float*)d_in[6];           // [H]
  const unsigned char* validm = (const unsigned char*)d_in[7]; // [B,N] bool
  const unsigned char* histm  = (const unsigned char*)d_in[8]; // [B,T] bool
  float* out = (float*)d_out;

  const int BT = 4 * T_DIM;                              // 16384 rows
  unsigned short* Qws = (unsigned short*)d_ws;           // BT*K bf16   (8 MB)
  unsigned short* Rws = Qws + (size_t)BT * K_DIM;        // BT*V bf16   (8 MB)
  unsigned short* Wqb = Rws + (size_t)BT * V_DIM;        // K*H bf16    (1 MB)
  unsigned short* Wob = Wqb + (size_t)K_DIM * H_DIM;     // H*V bf16    (1 MB)

  // one-shot weight conversion fp32 -> bf16
  k0_cvt<<<dim3(256), dim3(256), 0, stream>>>(Wq, Wqb, (K_DIM * H_DIM) / 4);
  k0_cvt<<<dim3(256), dim3(256), 0, stream>>>(Wo, Wob, (H_DIM * V_DIM) / 4);

  k1_qproj<<<dim3(BT / 16), dim3(256), 0, stream>>>(hid, Wqb, Qws);
  k2_attn <<<dim3(BT / 128), dim3(256), 0, stream>>>(Qws, keys, vals, validm, histm, Rws);
  k3_out  <<<dim3(BT / 16), dim3(256), 0, stream>>>(Rws, Wob, hid, gamma, beta, out);
}
